// PerformerBlock_8452495638914
// MI455X (gfx1250) — matrix operand order repaired and checked by reading, compiled, not run on hardware
//
#include <hip/hip_runtime.h>
#include <hip/hip_bf16.h>
#include <math.h>

// Problem constants (from reference)
#define BB    4
#define SS    4096
#define DDm   768
#define HHn   12
#define MF    256
#define DHd   64
#define DFF   3072
#define NTOK  (BB*SS)      // 16384
#define BH    (BB*HHn)     // 48
#define LN_EPS 1e-5f
#define EPS_K  1e-4f

typedef _Float16 v16h __attribute__((ext_vector_type(16)));
typedef _Float16 v8h  __attribute__((ext_vector_type(8)));
typedef float    v8f  __attribute__((ext_vector_type(8)));
typedef unsigned int v4u __attribute__((ext_vector_type(4)));
typedef int      v8i  __attribute__((ext_vector_type(8)));
typedef int      v4i  __attribute__((ext_vector_type(4)));

struct alignas(16) f16x8 { _Float16 v[8]; };

#if defined(__HIP_DEVICE_COMPILE__) && __has_builtin(__builtin_amdgcn_tensor_load_to_lds)
#define HAVE_TDM 1
#else
#define HAVE_TDM 0
#endif

// ---------------------------------------------------------------------------
// gfx1250 async global->LDS copy (ASYNCcnt-tracked DMA, no VGPR round trip).
// ---------------------------------------------------------------------------
__device__ __forceinline__ void async_ld_b128(void* lds, const void* gptr) {
  unsigned lds_addr = (unsigned)(size_t)lds;
  asm volatile("global_load_async_to_lds_b128 %0, %1, off"
               :: "v"(lds_addr), "v"(gptr) : "memory");
}
template<int N>
__device__ __forceinline__ void wait_async_le() {
  if      (N == 0) asm volatile("s_wait_asynccnt 0x0" ::: "memory");
  else if (N == 1) asm volatile("s_wait_asynccnt 0x1" ::: "memory");
  else if (N == 2) asm volatile("s_wait_asynccnt 0x2" ::: "memory");
  else             asm volatile("s_wait_asynccnt 0x3" ::: "memory");
}
__device__ __forceinline__ void wait_tensor0() {
#if __has_builtin(__builtin_amdgcn_s_wait_tensorcnt)
  __builtin_amdgcn_s_wait_tensorcnt(0);
#else
  asm volatile("s_wait_tensorcnt 0x0" ::: "memory");
#endif
}
__device__ __forceinline__ void wait_tensor1() {
#if __has_builtin(__builtin_amdgcn_s_wait_tensorcnt)
  __builtin_amdgcn_s_wait_tensorcnt(1);
#else
  asm volatile("s_wait_tensorcnt 0x1" ::: "memory");
#endif
}

// ---------------------------------------------------------------------------
// Tensor Data Mover: 2D f16 tile load, global -> LDS (one DMA descriptor).
// D# layout per CDNA5 ISA ch.8: group0 = {flags, lds_addr, global_addr, type},
// group1 = {mask/data_size/pad, tensor dims, tile dims, dim0 stride}.
// padI/padA are the encoded pad_interval / pad_amount codes.
// This toolchain's builtin is the 6-arg form:
//   (uint32x4 g0, int32x8 g1, int32x4 g2, int32x4 g3, int32x8 spare, i32 cpol)
// ---------------------------------------------------------------------------
__device__ __forceinline__ void tdm_load_2d_f16(void* lds, const void* gptr,
                                                int tdim0, int tdim1,
                                                int tile0, int tile1,
                                                long long stride0,
                                                int padI, int padA) {
#if HAVE_TDM
  unsigned long long ga = (unsigned long long)(size_t)gptr;
  v4u g0;
  g0.x = 1u;                                            // count=1 (user D#)
  g0.y = (unsigned)(size_t)lds;                         // lds_addr
  g0.z = (unsigned)(ga & 0xFFFFFFFFu);                  // global_addr[31:0]
  g0.w = (unsigned)((ga >> 32) & 0x01FFFFFFu) | (2u << 30);  // addr[56:32] | type=2
  unsigned w0 = (1u << 16);                             // data_size = 1 (2 bytes)
  if (padA > 0)
    w0 |= (1u << 20) | ((unsigned)padI << 22) | ((unsigned)padA << 25);
  v8i g1;
  g1[0] = (int)w0;
  g1[1] = (int)(((unsigned)tdim0 & 0xFFFFu) << 16);     // tensor_dim0[15:0]
  g1[2] = (int)(((unsigned)tdim0 >> 16) | (((unsigned)tdim1 & 0xFFFFu) << 16));
  g1[3] = (int)(((unsigned)tdim1 >> 16) | ((unsigned)tile0 << 16));
  g1[4] = (int)(unsigned)tile1;                         // tile_dim1 (tile_dim2=0)
  g1[5] = (int)(unsigned)(stride0 & 0xFFFFFFFFll);      // dim0_stride[31:0]
  g1[6] = (int)(unsigned)((stride0 >> 32) & 0xFFFFll);  // dim0_stride[47:32]
  g1[7] = 0;
  v4i zero4 = {0, 0, 0, 0};
  v8i zero8 = {0, 0, 0, 0, 0, 0, 0, 0};
  __builtin_amdgcn_tensor_load_to_lds(g0, g1, zero4, zero4, zero8, 0);
#else
  (void)lds; (void)gptr; (void)tdim0; (void)tdim1;
  (void)tile0; (void)tile1; (void)stride0; (void)padI; (void)padA;
#endif
}

// Assemble a v16h fragment from two 16B-aligned LDS chunks (2x ds_load_b128).
__device__ __forceinline__ v16h ld_frag(const _Float16* p0, const _Float16* p1) {
  v8h lo = *(const v8h*)p0;
  v8h hi = *(const v8h*)p1;
  return __builtin_shufflevector(lo, hi, 0,1,2,3,4,5,6,7,8,9,10,11,12,13,14,15);
}

// ---------------------------------------------------------------------------
// fp32 -> f16 weight conversion (grid-stride)
// ---------------------------------------------------------------------------
__global__ void cvt_f32_to_f16(const float* __restrict__ in, _Float16* __restrict__ out, int n) {
  int i = blockIdx.x * blockDim.x + threadIdx.x;
  int stride = gridDim.x * blockDim.x;
  for (; i < n; i += stride) out[i] = (_Float16)in[i];
}

// ---------------------------------------------------------------------------
// LayerNorm over D=768: one row per block, 256 threads x 3 elements
// ---------------------------------------------------------------------------
__global__ void ln_kernel(const float* __restrict__ x, const float* __restrict__ g,
                          const float* __restrict__ bb, _Float16* __restrict__ out) {
  __shared__ float s1[256], s2[256];
  int row = blockIdx.x, tid = threadIdx.x;
  const float* xr = x + (long long)row * DDm;
  float v0 = xr[tid], v1 = xr[tid + 256], v2 = xr[tid + 512];
  s1[tid] = v0 + v1 + v2;
  s2[tid] = v0*v0 + v1*v1 + v2*v2;
  __syncthreads();
  for (int o = 128; o > 0; o >>= 1) {
    if (tid < o) { s1[tid] += s1[tid + o]; s2[tid] += s2[tid + o]; }
    __syncthreads();
  }
  float mu  = s1[0] * (1.f / DDm);
  float var = s2[0] * (1.f / DDm) - mu * mu;
  float rstd = rsqrtf(var + LN_EPS);
  _Float16* orow = out + (long long)row * DDm;
  orow[tid]       = (_Float16)((v0 - mu) * rstd * g[tid]       + bb[tid]);
  orow[tid + 256] = (_Float16)((v1 - mu) * rstd * g[tid + 256] + bb[tid + 256]);
  orow[tid + 512] = (_Float16)((v2 - mu) * rstd * g[tid + 512] + bb[tid + 512]);
}

// ---------------------------------------------------------------------------
// Generic batched WMMA GEMM: C[i,j] = sum_k A[i,k] * B[k,j]  (+ epilogue)
//   TA=0: A row-major [Mrows x K] (lda)   TA=1: A stored [K x Mrows] (lda)
//   TB=0: B row-major [K x Ncols] (ldb)   TB=1: B stored [Ncols x K] (ldb)
//   EPI: 0=bias  1=bias+gelu  2=bias+residual(f32)  3=rowscale (d_inv)
// 256 threads (8 waves), block tile 128x64, K step 32, double-buffered LDS.
// Wave tile 32x32 = 2x2 v_wmma_f32_16x16x32_f16 with A/B fragment reuse.
// Staging: TA=0 -> per-lane async global->LDS DMA (ASYNCcnt);
//          TB=1 -> one TDM tensor_load_to_lds per tile (TENSORcnt, wave 0),
//                  LDS row padding 64B data + 16B pad via D# pad fields.
// Tile t+1 is staged into the alternate buffer while tile t computes.
// ---------------------------------------------------------------------------
template<int TA, int TB, int EPI>
__global__ void gemm_wmma(const _Float16* __restrict__ A, const _Float16* __restrict__ Bm,
                          const float* __restrict__ bias, const float* __restrict__ res,
                          const float* __restrict__ rowscale,
                          _Float16* __restrict__ outH, float* __restrict__ outF,
                          int Mrows, int Ncols, int K,
                          int lda, int ldb, int ldc, int ldres,
                          long long aOffB, long long aOffH,
                          long long bOffB, long long bOffH,
                          long long cOffB, long long cOffH,
                          long long rsStride, int Hn) {
  __shared__ __align__(16) _Float16 As[2][128][40];  // [buf][row][k], padded
  __shared__ __align__(16) _Float16 Bs[2][64][40];   // [buf][col][k], fragment-major
  int z = blockIdx.z;
  int bi = z / Hn, hi = z % Hn;
  const _Float16* bA = A  + bi * aOffB + hi * aOffH;
  const _Float16* bB = Bm + bi * bOffB + hi * bOffH;
  long long cOff = bi * cOffB + hi * cOffH;

  int tid  = threadIdx.x;
  int wave = tid >> 5, lane = tid & 31;
  int half = lane >> 4, l16 = lane & 15;
  int rw = wave >> 1, cw = wave & 1;             // 4x2 wave grid of 32x32 tiles
  int row0 = blockIdx.y * 128, n0 = blockIdx.x * 64;

  auto stage = [&](int p, int k0) {
    if (TA == 0) {
      int r = tid >> 1, kseg = (tid & 1) * 16;
      const _Float16* gp = &bA[(long long)(row0 + r) * lda + k0 + kseg];
      async_ld_b128(&As[p][r][kseg],     gp);
      async_ld_b128(&As[p][r][kseg + 8], gp + 8);
      if (k0 + 32 < K)  // gfx1250 global_prefetch_b8 of the following k-tile
        __builtin_prefetch(gp + 32, 0, 1);
    } else {
      int kk = tid >> 3, rseg = (tid & 7) * 16;
      f16x8 t0 = *(const f16x8*)&bA[(long long)(k0 + kk) * lda + row0 + rseg];
      f16x8 t1 = *(const f16x8*)&bA[(long long)(k0 + kk) * lda + row0 + rseg + 8];
      #pragma unroll
      for (int j = 0; j < 8; ++j) {
        As[p][rseg + j][kk] = t0.v[j];
        As[p][rseg + 8 + j][kk] = t1.v[j];
      }
    }
    if (TB == 1) {
#if HAVE_TDM
      if (wave == 0) {  // one TDM DMA for the whole 64x32 B tile
        tdm_load_2d_f16(&Bs[p][0][0], &bB[(long long)n0 * ldb + k0],
                        K - k0, Ncols - n0, /*tile*/32, 64,
                        (long long)ldb, /*padI: 16 dw*/3, /*padA: 4 dw*/3);
      }
#else
      int nn = tid >> 2, kseg = (tid & 3) * 8;
      async_ld_b128(&Bs[p][nn][kseg], &bB[(long long)(n0 + nn) * ldb + k0 + kseg]);
#endif
    } else {
      int kk = tid >> 3, nseg = (tid & 7) * 8;
      f16x8 tmp = *(const f16x8*)&bB[(long long)(k0 + kk) * ldb + n0 + nseg];
      #pragma unroll
      for (int j = 0; j < 8; ++j) Bs[p][nseg + j][kk] = tmp.v[j];
    }
  };

  v8f acc[2][2];
  #pragma unroll
  for (int si = 0; si < 2; ++si)
    #pragma unroll
    for (int sj = 0; sj < 2; ++sj)
      acc[si][sj] = (v8f){0.f,0.f,0.f,0.f,0.f,0.f,0.f,0.f};

  constexpr int PEND = (TA == 0 ? 2 : 0) + ((TB == 1 && !HAVE_TDM) ? 1 : 0);

  stage(0, 0);                                  // prologue: tile 0
  int nsteps = K >> 5;
  for (int t = 0; t < nsteps; ++t) {
    int p = t & 1;
    bool hasNext = (t + 1 < nsteps);
    if (hasNext) stage(1 - p, (t + 1) << 5);    // DMA tile t+1 overlaps tile t math
    if (PEND > 0) { if (hasNext) wait_async_le<PEND>(); else wait_async_le<0>(); }
    if (TB == 1 && HAVE_TDM && wave == 0) { if (hasNext) wait_tensor1(); else wait_tensor0(); }
    __syncthreads();                            // tile t fully in LDS for all waves

    v16h a[2], b[2];
    #pragma unroll
    for (int si = 0; si < 2; ++si) {
      int ar = rw * 32 + si * 16 + l16;
      a[si] = ld_frag(&As[p][ar][half * 8], &As[p][ar][16 + half * 8]);
    }
    #pragma unroll
    for (int sj = 0; sj < 2; ++sj) {
      int bc = cw * 32 + sj * 16 + l16;
      b[sj] = ld_frag(&Bs[p][bc][half * 8], &Bs[p][bc][16 + half * 8]);
    }
    #pragma unroll
    for (int si = 0; si < 2; ++si)
      #pragma unroll
      for (int sj = 0; sj < 2; ++sj)
        acc[si][sj] = __builtin_amdgcn_wmma_f32_16x16x32_f16(
            false, a[si], false, b[sj], (short)0, acc[si][sj], false, false);
    __syncthreads();                            // all reads of buf p done
  }

  // ---- epilogue: C layout VGPR i -> row half*8+i, lanes -> cols ----
  #pragma unroll
  for (int si = 0; si < 2; ++si) {
    #pragma unroll
    for (int sj = 0; sj < 2; ++sj) {
      #pragma unroll
      for (int i = 0; i < 8; ++i) {
        int rl = rw * 32 + si * 16 + half * 8 + i;
        int cl = cw * 32 + sj * 16 + l16;
        int gr = row0 + rl, gc = n0 + cl;
        if (gr < Mrows && gc < Ncols) {
          float v = acc[si][sj][i];
          if (EPI != 3 && bias) v += bias[gc];
          if (EPI == 1) v = 0.5f * v * (1.f + erff(v * 0.70710678118f));  // exact gelu
          if (EPI == 2) v += res[(long long)gr * ldres + gc];
          if (EPI == 3) v *= rowscale[rsStride * z + gr];
          long long ci = cOff + (long long)gr * ldc + gc;
          if (outH) outH[ci] = (_Float16)v;
          if (outF) outF[ci] = v;
        }
      }
    }
  }
}

// ---------------------------------------------------------------------------
// FAVOR+ feature map: dd = (q*norm) @ proj^T per head; then
//   MODE 0 (query): qp = ratio*(exp(dd - diag - rowmax(dd)) + eps)
//   MODE 1 (key, phase A): global atomic max of dd
//   MODE 2 (key, phase B): kp = ratio*(exp(dd - diag - kmax) + eps)
// Block: 64 rows x full M=256 cols; proj tile via one TDM DMA (144B padded
// rows), q tile via per-lane async DMA; fragments contiguous ds_load_b128.
// ---------------------------------------------------------------------------
template<int MODE>
__global__ void feature_kernel(const _Float16* __restrict__ qk,
                               const _Float16* __restrict__ proj,
                               float* __restrict__ kmax, _Float16* __restrict__ outP,
                               float norm, float ratio) {
  __shared__ __align__(16) _Float16 Qs[64][72];
  __shared__ __align__(16) _Float16 Ps[MF][72];
  __shared__ float diag[64];
  __shared__ float part[2][64];
  __shared__ float wmaxs[8];

  int z = blockIdx.z;
  int bi = z / HHn, hi = z % HHn;
  const _Float16* bQ = qk + (long long)bi * SS * DDm + hi * DHd;
  int row0 = blockIdx.x * 64;
  int tid = threadIdx.x;
  int wave = tid >> 5, lane = tid & 31;

  { // async-stage 64x64 q tile
    int r = tid >> 2, dseg = (tid & 3) * 16;
    const _Float16* gp = &bQ[(long long)(row0 + r) * DDm + dseg];
    async_ld_b128(&Qs[r][dseg],     gp);
    async_ld_b128(&Qs[r][dseg + 8], gp + 8);
  }
#if HAVE_TDM
  if (wave == 0) {  // one TDM DMA for the whole 256x64 proj tile
    tdm_load_2d_f16(&Ps[0][0], proj, DHd, MF, /*tile*/DHd, MF,
                    (long long)DHd, /*padI: 32 dw*/4, /*padA: 4 dw*/3);
  }
#else
  { const _Float16* pr = proj + tid * DHd;
    #pragma unroll
    for (int j = 0; j < 8; ++j) async_ld_b128(&Ps[tid][j * 8], &pr[j * 8]); }
#endif
  wait_async_le<0>();
#if HAVE_TDM
  if (wave == 0) wait_tensor0();
#endif
  __syncthreads();

  if (MODE != 1 && tid < 64) {               // diag = 0.5 * norm^2 * sum(q^2)
    float s = 0.f;
    #pragma unroll 8
    for (int d = 0; d < DHd; ++d) { float q = (float)Qs[tid][d]; s += q * q; }
    diag[tid] = s * 0.5f * norm * norm;
  }

  int half = lane >> 4, l16 = lane & 15;
  int rt = wave >> 1, cw = wave & 1;

  // A fragments for this wave's 16 rows (K=64 -> 2 steps of 32)
  int ar = rt * 16 + l16;
  v16h a0 = ld_frag(&Qs[ar][half * 8],      &Qs[ar][16 + half * 8]);
  v16h a1 = ld_frag(&Qs[ar][32 + half * 8], &Qs[ar][48 + half * 8]);

  float vals[8][8];
  float rmax[8];
  #pragma unroll
  for (int i = 0; i < 8; ++i) rmax[i] = -3.0e38f;

  #pragma unroll
  for (int t = 0; t < 8; ++t) {              // 8 feature-col tiles per wave
    int mrow = (cw * 8 + t) * 16 + l16;      // feature index m = B-matrix col
    v16h b0 = ld_frag(&Ps[mrow][half * 8],      &Ps[mrow][16 + half * 8]);
    v16h b1 = ld_frag(&Ps[mrow][32 + half * 8], &Ps[mrow][48 + half * 8]);
    v8f acc = {0.f,0.f,0.f,0.f,0.f,0.f,0.f,0.f};
    acc = __builtin_amdgcn_wmma_f32_16x16x32_f16(false, a0, false, b0, (short)0, acc, false, false);
    acc = __builtin_amdgcn_wmma_f32_16x16x32_f16(false, a1, false, b1, (short)0, acc, false, false);
    #pragma unroll
    for (int i = 0; i < 8; ++i) {
      float s = acc[i] * norm;
      vals[t][i] = s;
      rmax[i] = fmaxf(rmax[i], s);
    }
  }

  if (MODE == 1) {                           // block max -> global atomic max
    float m = -3.0e38f;
    #pragma unroll
    for (int i = 0; i < 8; ++i) m = fmaxf(m, rmax[i]);
    #pragma unroll
    for (int o = 16; o > 0; o >>= 1) m = fmaxf(m, __shfl_xor(m, o, 32));
    if (lane == 0) wmaxs[wave] = m;
    __syncthreads();
    if (tid == 0) {
      float bm = wmaxs[0];
      for (int w = 1; w < 8; ++w) bm = fmaxf(bm, wmaxs[w]);
      int* addr = (int*)kmax;                // CAS float max (values can be <0)
      int old = *addr;
      while (bm > __int_as_float(old)) {
        int assumed = old;
        old = atomicCAS(addr, assumed, __float_as_int(bm));
        if (old == assumed) break;
      }
    }
    return;
  }

  if (MODE == 0) {                           // per-row max within each 16-lane half
    #pragma unroll
    for (int i = 0; i < 8; ++i) {
      float m = rmax[i];
      m = fmaxf(m, __shfl_xor(m, 1, 32));
      m = fmaxf(m, __shfl_xor(m, 2, 32));
      m = fmaxf(m, __shfl_xor(m, 4, 32));
      m = fmaxf(m, __shfl_xor(m, 8, 32));
      if (l16 == 0) part[cw][rt * 16 + half * 8 + i] = m;
    }
  }
  __syncthreads();

  float kmx = 0.f;
  if (MODE == 2) kmx = *kmax;
  _Float16* op = outP + (long long)z * SS * MF + (long long)row0 * MF;
  #pragma unroll
  for (int t = 0; t < 8; ++t) {
    #pragma unroll
    for (int i = 0; i < 8; ++i) {
      int rl = rt * 16 + half * 8 + i;
      int cl = (cw * 8 + t) * 16 + l16;
      float mx = (MODE == 0) ? fmaxf(part[0][rl], part[1][rl]) : kmx;
      float v = ratio * (expf(vals[t][i] - diag[rl] - mx) + EPS_K);
      op[(long long)rl * MF + cl] = (_Float16)v;
    }
  }
}

// ---------------------------------------------------------------------------
// k_sum[bh][m] = sum_n kp[bh][n][m]   (coalesced over m)
// ---------------------------------------------------------------------------
__global__ void ksum_kernel(const _Float16* __restrict__ kp, float* __restrict__ ksum) {
  int z = blockIdx.x, m = threadIdx.x;
  const _Float16* base = kp + (long long)z * SS * MF;
  float s = 0.f;
  for (int n = 0; n < SS; ++n) s += (float)base[(long long)n * MF + m];
  ksum[z * MF + m] = s;
}

// ---------------------------------------------------------------------------
// d_inv[bh][n] = 1 / (qp[bh][n][:] . k_sum[bh][:])
// ---------------------------------------------------------------------------
__global__ void dinv_kernel(const _Float16* __restrict__ qp, const float* __restrict__ ksum,
                            float* __restrict__ dinv) {
  __shared__ float ks[MF];
  int z = blockIdx.x, tid = threadIdx.x;
  ks[tid] = ksum[z * MF + tid];
  __syncthreads();
  int n = blockIdx.y * 256 + tid;
  const _Float16* row = qp + (long long)z * SS * MF + (long long)n * MF;
  float s = 0.f;
  #pragma unroll 8
  for (int m = 0; m < MF; ++m) s += (float)row[m] * ks[m];
  dinv[z * SS + n] = 1.f / s;
}

__global__ void init_kmax(float* kmax) {
  if (threadIdx.x == 0 && blockIdx.x == 0) *kmax = -3.0e38f;
}

// ---------------------------------------------------------------------------
extern "C" void kernel_launch(void* const* d_in, const int* in_sizes, int n_in,
                              void* d_out, int out_size, void* d_ws, size_t ws_size,
                              hipStream_t stream) {
  const float* x    = (const float*)d_in[0];
  const float* proj = (const float*)d_in[1];
  const float* wq = (const float*)d_in[2];  const float* bq = (const float*)d_in[3];
  const float* wk = (const float*)d_in[4];  const float* bk = (const float*)d_in[5];
  const float* wv = (const float*)d_in[6];  const float* bv = (const float*)d_in[7];
  const float* wo = (const float*)d_in[8];  const float* bo = (const float*)d_in[9];
  const float* ln1g = (const float*)d_in[10]; const float* ln1b = (const float*)d_in[11];
  const float* ln2g = (const float*)d_in[12]; const float* ln2b = (const float*)d_in[13];
  const float* w1 = (const float*)d_in[14]; const float* b1 = (const float*)d_in[15];
  const float* w2 = (const float*)d_in[16]; const float* b2 = (const float*)d_in[17];
  float* out = (float*)d_out;

  // ---- workspace carve-up (256B aligned; total ~320 MB with aliasing) ----
  char* ws = (char*)d_ws;
  size_t off = 0;
  auto alloc = [&](size_t bytes) -> void* {
    void* p = ws + off;
    off = (off + bytes + 255) & ~(size_t)255;
    return p;
  };
  _Float16* wq16   = (_Float16*)alloc((size_t)DDm * DDm * 2);
  _Float16* wk16   = (_Float16*)alloc((size_t)DDm * DDm * 2);
  _Float16* wv16   = (_Float16*)alloc((size_t)DDm * DDm * 2);
  _Float16* wo16   = (_Float16*)alloc((size_t)DDm * DDm * 2);
  _Float16* w116   = (_Float16*)alloc((size_t)DFF * DDm * 2);
  _Float16* w216   = (_Float16*)alloc((size_t)DDm * DFF * 2);
  _Float16* proj16 = (_Float16*)alloc((size_t)MF * DHd * 2);
  _Float16* h16    = (_Float16*)alloc((size_t)NTOK * DDm * 2);
  _Float16* q16    = (_Float16*)alloc((size_t)NTOK * DDm * 2);
  _Float16* k16    = (_Float16*)alloc((size_t)NTOK * DDm * 2);
  _Float16* v16    = (_Float16*)alloc((size_t)NTOK * DDm * 2);
  _Float16* qp     = (_Float16*)alloc((size_t)BH * SS * MF * 2);
  _Float16* kp     = (_Float16*)alloc((size_t)BH * SS * MF * 2);
  _Float16* ctx16  = (_Float16*)alloc((size_t)BH * MF * DHd * 2);
  float*    ksumB  = (float*)alloc((size_t)BH * MF * 4);
  float*    dinvB  = (float*)alloc((size_t)BH * SS * 4);
  float*    kmaxB  = (float*)alloc(256);
  // lifetime-based aliases (regions no longer live when reused):
  _Float16* attn16 = q16;               // q dead after qp computed
  float*    x2     = (float*)k16;       // spans k16+v16 (2*25 MB = 50 MB f32)
  _Float16* h2_16  = kp;                // kp dead after ctx
  _Float16* g16    = qp;                // qp dead after attn
  (void)in_sizes; (void)n_in; (void)out_size; (void)ws_size;

  // ---- 1. convert weights / proj to f16 ----
  cvt_f32_to_f16<<<1024, 256, 0, stream>>>(wq, wq16, DDm * DDm);
  cvt_f32_to_f16<<<1024, 256, 0, stream>>>(wk, wk16, DDm * DDm);
  cvt_f32_to_f16<<<1024, 256, 0, stream>>>(wv, wv16, DDm * DDm);
  cvt_f32_to_f16<<<1024, 256, 0, stream>>>(wo, wo16, DDm * DDm);
  cvt_f32_to_f16<<<1024, 256, 0, stream>>>(w1, w116, DFF * DDm);
  cvt_f32_to_f16<<<1024, 256, 0, stream>>>(w2, w216, DDm * DFF);
  cvt_f32_to_f16<<<64, 256, 0, stream>>>(proj, proj16, MF * DHd);

  // ---- 2. LN1 ----
  ln_kernel<<<NTOK, 256, 0, stream>>>(x, ln1g, ln1b, h16);

  // ---- 3-5. QKV projections: [16384x768] = h @ W^T + b ----
  dim3 gQKV(DDm / 64, NTOK / 128, 1);
  gemm_wmma<0, 1, 0><<<gQKV, 256, 0, stream>>>(h16, wq16, bq, nullptr, nullptr, q16, nullptr,
      NTOK, DDm, DDm, DDm, DDm, DDm, 0, 0, 0, 0, 0, 0, 0, 0, 1);
  gemm_wmma<0, 1, 0><<<gQKV, 256, 0, stream>>>(h16, wk16, bk, nullptr, nullptr, k16, nullptr,
      NTOK, DDm, DDm, DDm, DDm, DDm, 0, 0, 0, 0, 0, 0, 0, 0, 1);
  gemm_wmma<0, 1, 0><<<gQKV, 256, 0, stream>>>(h16, wv16, bv, nullptr, nullptr, v16, nullptr,
      NTOK, DDm, DDm, DDm, DDm, DDm, 0, 0, 0, 0, 0, 0, 0, 0, 1);

  // ---- 6-9. FAVOR+ feature maps ----
  float norm  = powf((float)DHd, -0.25f);
  float ratio = rsqrtf((float)MF);
  dim3 gF(SS / 64, 1, BH);
  feature_kernel<0><<<gF, 256, 0, stream>>>(q16, proj16, nullptr, qp, norm, ratio);
  init_kmax<<<1, 64, 0, stream>>>(kmaxB);
  feature_kernel<1><<<gF, 256, 0, stream>>>(k16, proj16, kmaxB, nullptr, norm, ratio);
  feature_kernel<2><<<gF, 256, 0, stream>>>(k16, proj16, kmaxB, kp, norm, ratio);

  // ---- 10. k_sum ----
  ksum_kernel<<<BH, MF, 0, stream>>>(kp, ksumB);

  // ---- 11. ctx[bh][m][d] = kp^T v : TA=1 (A stored [S x M]), TB=0 (v [S x D] slice) ----
  dim3 gCtx(DHd / 64, MF / 128, BH);
  gemm_wmma<1, 0, 0><<<gCtx, 256, 0, stream>>>(kp, v16, nullptr, nullptr, nullptr, ctx16, nullptr,
      MF, DHd, SS, MF, DDm, DHd, 0,
      (long long)HHn * SS * MF, (long long)SS * MF,
      (long long)SS * DDm, (long long)DHd,
      (long long)HHn * MF * DHd, (long long)MF * DHd, 0, HHn);

  // ---- 12. d_inv ----
  dinv_kernel<<<dim3(BH, SS / 256), 256, 0, stream>>>(qp, ksumB, dinvB);

  // ---- 13. attn[bh][n][d] = d_inv[n] * (qp @ ctx), scattered to [bs, h*64+d] ----
  dim3 gAttn(DHd / 64, SS / 128, BH);
  gemm_wmma<0, 0, 3><<<gAttn, 256, 0, stream>>>(qp, ctx16, nullptr, nullptr, dinvB, attn16, nullptr,
      SS, DHd, MF, MF, DHd, DDm, 0,
      (long long)HHn * SS * MF, (long long)SS * MF,
      (long long)HHn * MF * DHd, (long long)MF * DHd,
      (long long)SS * DDm, (long long)DHd, (long long)SS, HHn);

  // ---- 14. out-projection + residual: x2 = x + attn @ wo^T + bo ----
  gemm_wmma<0, 1, 2><<<gQKV, 256, 0, stream>>>(attn16, wo16, bo, x, nullptr, nullptr, x2,
      NTOK, DDm, DDm, DDm, DDm, DDm, DDm, 0, 0, 0, 0, 0, 0, 0, 1);

  // ---- 15. LN2 ----
  ln_kernel<<<NTOK, 256, 0, stream>>>(x2, ln2g, ln2b, h2_16);

  // ---- 16. FFN1 + exact gelu ----
  dim3 gFF1(DFF / 64, NTOK / 128, 1);
  gemm_wmma<0, 1, 1><<<gFF1, 256, 0, stream>>>(h2_16, w116, b1, nullptr, nullptr, g16, nullptr,
      NTOK, DFF, DDm, DDm, DDm, DFF, 0, 0, 0, 0, 0, 0, 0, 0, 1);

  // ---- 17. FFN2 + residual -> d_out (f32) ----
  gemm_wmma<0, 1, 2><<<gQKV, 256, 0, stream>>>(g16, w216, b2, x2, nullptr, nullptr, out,
      NTOK, DDm, DFF, DFF, DFF, DDm, DDm, 0, 0, 0, 0, 0, 0, 0, 1);
}
